// LlamaDecoderLayer_50483045597485
// MI455X (gfx1250) — matrix operand order repaired and checked by reading, not compiled, not run on hardware
//
#include <hip/hip_runtime.h>
#include <math.h>

typedef __attribute__((ext_vector_type(16))) _Float16 v16h;
typedef __attribute__((ext_vector_type(8)))  _Float16 v8h;
typedef __attribute__((ext_vector_type(8)))  float    v8f;

#define SEQ   4096
#define HID   2048
#define NH    16
#define NKV   4
#define HD    128
#define NQKV  3072          // (16 + 2*4) * 128
#define INTER 5632
#define GUP   (2*INTER)

__device__ __forceinline__ v16h pack16(v8h lo, v8h hi) {
  v16h r;
#pragma unroll
  for (int i = 0; i < 8; ++i) { r[i] = lo[i]; r[i + 8] = hi[i]; }
  return r;
}

__device__ __forceinline__ v8f wmma_f16(v16h a, v16h b, v8f c) {
  // D = A(16x32 f16) x B(32x16 f16) + C(16x16 f32)
  return __builtin_amdgcn_wmma_f32_16x16x32_f16(false, a, false, b, (short)0, c,
                                                false, false);
}

// ---------------------------------------------------------------------------
// RMSNorm (f32 in) -> f16 out
// ---------------------------------------------------------------------------
__global__ void rmsnorm_f16_kernel(const float* __restrict__ x,
                                   const float* __restrict__ w,
                                   _Float16* __restrict__ y, int cols) {
  __shared__ float red[256];
  int row = blockIdx.x;
  const float* xr = x + (size_t)row * cols;
  float s = 0.f;
  for (int j = threadIdx.x; j < cols; j += 256) { float v = xr[j]; s += v * v; }
  red[threadIdx.x] = s;
  __syncthreads();
  for (int off = 128; off; off >>= 1) {
    if ((int)threadIdx.x < off) red[threadIdx.x] += red[threadIdx.x + off];
    __syncthreads();
  }
  float r = rsqrtf(red[0] / (float)cols + 1e-5f);
  _Float16* yr = y + (size_t)row * cols;
  for (int j = threadIdx.x; j < cols; j += 256)
    yr[j] = (_Float16)(xr[j] * r * w[j]);
}

// ---------------------------------------------------------------------------
// One-time weight prep: f32 [K][N] row-major  ->  f16 [N][K] row-major
// ---------------------------------------------------------------------------
__global__ void transpose_f32_to_f16(const float* __restrict__ src,
                                     _Float16* __restrict__ dst,
                                     int K, int N) {
  __shared__ float tile[32][33];
  int n0 = blockIdx.x * 32, k0 = blockIdx.y * 32;
#pragma unroll
  for (int i = 0; i < 4; ++i)
    tile[threadIdx.y + i * 8][threadIdx.x] =
        src[(size_t)(k0 + threadIdx.y + i * 8) * N + n0 + threadIdx.x];
  __syncthreads();
#pragma unroll
  for (int i = 0; i < 4; ++i)
    dst[(size_t)(n0 + threadIdx.y + i * 8) * K + k0 + threadIdx.x] =
        (_Float16)tile[threadIdx.x][threadIdx.y + i * 8];
}

// ---------------------------------------------------------------------------
// WMMA GEMM: C[M][N] (f32) = A[M][K] (f16) * BT[N][K]^T (f16) (+ residual)
// wave: 32x64 tile (2 A-frags x 4 B-frags = 8 accumulators, B reused 2x);
// block: 8 waves -> 32 x 512 per workgroup.
// ---------------------------------------------------------------------------
__global__ void __launch_bounds__(256)
gemm_wmma_f16(const _Float16* __restrict__ A, const _Float16* __restrict__ BT,
              const float* __restrict__ residual, float* __restrict__ C,
              int M, int N, int K) {
  int lane = threadIdx.x & 31;
  int wid  = threadIdx.x >> 5;
  int l15  = lane & 15;
  int hi   = lane >> 4;
  int m0   = blockIdx.x * 32;
  int c0   = blockIdx.y * 512 + wid * 64;

  const _Float16* arow0 = A + (size_t)(m0 + l15) * K;
  const _Float16* arow1 = A + (size_t)(m0 + 16 + l15) * K;
  v8f acc[2][4] = {};
  for (int k0 = 0; k0 < K; k0 += 32) {
    // A fragments (16x32): lo lanes hold K{k0..k0+15}, hi lanes +16
    v16h a0 = pack16(*(const v8h*)(arow0 + k0 + hi * 16),
                     *(const v8h*)(arow0 + k0 + 8 + hi * 16));
    v16h a1 = pack16(*(const v8h*)(arow1 + k0 + hi * 16),
                     *(const v8h*)(arow1 + k0 + 8 + hi * 16));
#pragma unroll
    for (int nt = 0; nt < 4; ++nt) {
      // B fragment (32x16): lane column n, lo lanes K{k0..k0+15}, hi +16
      const _Float16* bp =
          BT + (size_t)(c0 + nt * 16 + l15) * K + k0 + hi * 16;
      v16h b = *(const v16h*)bp;
      acc[0][nt] = wmma_f16(a0, b, acc[0][nt]);
      acc[1][nt] = wmma_f16(a1, b, acc[1][nt]);
    }
  }
#pragma unroll
  for (int mi = 0; mi < 2; ++mi) {
#pragma unroll
    for (int nt = 0; nt < 4; ++nt) {
#pragma unroll
      for (int r = 0; r < 8; ++r) {
        size_t idx = (size_t)(m0 + mi * 16 + r + hi * 8) * N +
                     (c0 + nt * 16 + l15);
        float v = acc[mi][nt][r];
        if (residual) v += residual[idx];
        C[idx] = v;
      }
    }
  }
}

// ---------------------------------------------------------------------------
// RoPE + split QKV. q scaled by HEAD_DIM^-0.5. V stored transposed [kvh][d][s].
// ---------------------------------------------------------------------------
__global__ void rope_split_kernel(const int* __restrict__ pos,
                                  const float* __restrict__ qkv,
                                  _Float16* __restrict__ q16,
                                  _Float16* __restrict__ k16,
                                  _Float16* __restrict__ vT16) {
  int s = blockIdx.x;
  float p = (float)pos[s];
  const float* row = qkv + (size_t)s * NQKV;
  const float qscale = 0.08838834764831845f;  // 128^-0.5
  const float ln_theta = 9.210340371976184f;  // ln(10000)

  for (int i = threadIdx.x; i < NH * 64; i += blockDim.x) {
    int h = i >> 6, j = i & 63;
    float x1 = row[h * HD + j], x2 = row[h * HD + j + 64];
    float f = p * __expf(-(float)(2 * j) * (1.0f / HD) * ln_theta);
    float cs, sn;
    __sincosf(f, &sn, &cs);
    _Float16* dst = q16 + ((size_t)s * NH + h) * HD;
    dst[j]      = (_Float16)((x1 * cs - x2 * sn) * qscale);
    dst[j + 64] = (_Float16)((x2 * cs + x1 * sn) * qscale);
  }
  for (int i = threadIdx.x; i < NKV * 64; i += blockDim.x) {
    int h = i >> 6, j = i & 63;
    const float* kr = row + NH * HD;
    float x1 = kr[h * HD + j], x2 = kr[h * HD + j + 64];
    float f = p * __expf(-(float)(2 * j) * (1.0f / HD) * ln_theta);
    float cs, sn;
    __sincosf(f, &sn, &cs);
    _Float16* dst = k16 + ((size_t)s * NKV + h) * HD;
    dst[j]      = (_Float16)(x1 * cs - x2 * sn);
    dst[j + 64] = (_Float16)(x2 * cs + x1 * sn);
  }
  for (int i = threadIdx.x; i < NKV * HD; i += blockDim.x) {
    int h = i >> 7, d = i & 127;
    float v = row[(NH + NKV) * HD + h * HD + d];
    vT16[((size_t)h * HD + d) * SEQ + s] = (_Float16)v;
  }
}

// ---------------------------------------------------------------------------
// Flash attention: 1 wave per (16-query block, head). Causal, GQA (4 KV heads).
// Per 32-key step: 8 WMMAs (S = Q*K^T) + online softmax + 8 WMMAs (O += P*V).
// ---------------------------------------------------------------------------
__global__ void __launch_bounds__(32)
flash_attn_kernel(const _Float16* __restrict__ q16,
                  const _Float16* __restrict__ k16,
                  const _Float16* __restrict__ vT16,
                  _Float16* __restrict__ attn16) {
  __shared__ alignas(64) _Float16 pl[16 * 32];  // P tile staging (C->A layout)
  int lane = threadIdx.x;
  int l15 = lane & 15, hi = lane >> 4;
  int qb = blockIdx.x, h = blockIdx.y, kvh = h >> 2;

  // Q fragments: 16 rows x 128, four K-chunks of 32
  const _Float16* qp = q16 + ((size_t)(qb * 16 + l15) * NH + h) * HD;
  v16h qf[4];
#pragma unroll
  for (int c = 0; c < 4; ++c) {
    v8h lo = *(const v8h*)(qp + c * 32 + hi * 16);
    v8h hh = *(const v8h*)(qp + c * 32 + 8 + hi * 16);
    qf[c] = pack16(lo, hh);
  }

  v8f o[8] = {};
  float m[8], l[8];
#pragma unroll
  for (int r = 0; r < 8; ++r) { m[r] = -__builtin_inff(); l[r] = 0.f; }

  int kend = qb * 16 + 15;
  for (int kbase = 0; kbase <= kend; kbase += 32) {
    v8f s0 = {}, s1 = {};
#pragma unroll
    for (int c = 0; c < 4; ++c) {
      v16h b0 = *(const v16h*)(k16 + ((size_t)(kbase + l15) * NKV + kvh) * HD +
                               c * 32 + hi * 16);
      s0 = wmma_f16(qf[c], b0, s0);
      v16h b1 = *(const v16h*)(k16 +
                               ((size_t)(kbase + 16 + l15) * NKV + kvh) * HD +
                               c * 32 + hi * 16);
      s1 = wmma_f16(qf[c], b1, s1);
    }
    // online softmax per row (row M = r + hi*8, values spread over 16 lanes)
#pragma unroll
    for (int r = 0; r < 8; ++r) {
      int row = qb * 16 + r + hi * 8;
      float v0 = (kbase + l15 <= row)      ? s0[r] : -__builtin_inff();
      float v1 = (kbase + 16 + l15 <= row) ? s1[r] : -__builtin_inff();
      float mx = fmaxf(v0, v1);
#pragma unroll
      for (int off = 8; off; off >>= 1) mx = fmaxf(mx, __shfl_xor(mx, off, 32));
      float mn = fmaxf(m[r], mx);
      float sc = __expf(m[r] - mn);
      float p0 = __expf(v0 - mn);
      float p1 = __expf(v1 - mn);
      float ps = p0 + p1;
#pragma unroll
      for (int off = 8; off; off >>= 1) ps += __shfl_xor(ps, off, 32);
      l[r] = l[r] * sc + ps;
      m[r] = mn;
#pragma unroll
      for (int c = 0; c < 8; ++c) o[c][r] = o[c][r] * sc;
      pl[(r + hi * 8) * 32 + l15]      = (_Float16)p0;
      pl[(r + hi * 8) * 32 + 16 + l15] = (_Float16)p1;
    }
    asm volatile("s_wait_dscnt 0x0" ::: "memory");
    // P as A-fragment (16x32)
    v8h plo = *(const v8h*)(pl + l15 * 32 + hi * 16);
    v8h phi = *(const v8h*)(pl + l15 * 32 + 8 + hi * 16);
    v16h pa = pack16(plo, phi);
    // O(16x128) += P(16x32) * V(32x128); V is pre-transposed [kvh][d][seq]
#pragma unroll
    for (int c = 0; c < 8; ++c) {
      v16h vb = *(const v16h*)(vT16 + ((size_t)kvh * HD + c * 16 + l15) * SEQ +
                               kbase + hi * 16);
      o[c] = wmma_f16(pa, vb, o[c]);
    }
    asm volatile("s_wait_dscnt 0x0" ::: "memory");
  }
  // epilogue: O / l  -> f16 attn buffer [s][2048]
#pragma unroll
  for (int r = 0; r < 8; ++r) {
    float inv = 1.0f / l[r];
    size_t rowoff = (size_t)(qb * 16 + r + hi * 8) * (NH * HD) + h * HD;
#pragma unroll
    for (int c = 0; c < 8; ++c)
      attn16[rowoff + c * 16 + l15] = (_Float16)(o[c][r] * inv);
  }
}

// ---------------------------------------------------------------------------
// act = silu(gate) * up  (f32 gate_up [S][2*INTER] -> f16 [S][INTER])
// ---------------------------------------------------------------------------
__global__ void silu_mul_kernel(const float* __restrict__ gu,
                                _Float16* __restrict__ act) {
  size_t i = (size_t)blockIdx.x * blockDim.x + threadIdx.x;
  size_t total = (size_t)SEQ * INTER;
  if (i >= total) return;
  size_t s = i / INTER, j = i % INTER;
  float g = gu[s * GUP + j];
  float u = gu[s * GUP + INTER + j];
  float si = g / (1.0f + __expf(-g));
  act[i] = (_Float16)(si * u);
}

// ---------------------------------------------------------------------------
extern "C" void kernel_launch(void* const* d_in, const int* in_sizes, int n_in,
                              void* d_out, int out_size, void* d_ws,
                              size_t ws_size, hipStream_t stream) {
  const int*   positions = (const int*)d_in[0];
  const float* hidden    = (const float*)d_in[1];
  const float* ln1_w     = (const float*)d_in[2];
  const float* ln2_w     = (const float*)d_in[3];
  const float* w_qkv     = (const float*)d_in[4];
  const float* w_o       = (const float*)d_in[5];
  const float* w_gate_up = (const float*)d_in[6];
  const float* w_down    = (const float*)d_in[7];
  float* out = (float*)d_out;
  (void)in_sizes; (void)n_in; (void)out_size; (void)ws_size;

  char* ws = (char*)d_ws;
  size_t off = 0;
  auto alloc = [&](size_t bytes) {
    char* p = ws + off;
    off = (off + bytes + 255) & ~(size_t)255;
    return p;
  };
  _Float16* h1n    = (_Float16*)alloc((size_t)SEQ * HID * 2);
  _Float16* wqkvT  = (_Float16*)alloc((size_t)NQKV * HID * 2);
  _Float16* woT    = (_Float16*)alloc((size_t)HID * HID * 2);
  _Float16* wguT   = (_Float16*)alloc((size_t)GUP * HID * 2);
  _Float16* wdnT   = (_Float16*)alloc((size_t)HID * INTER * 2);
  float*    qkv    = (float*)alloc((size_t)SEQ * NQKV * 4);
  _Float16* q16    = (_Float16*)alloc((size_t)SEQ * NH * HD * 2);
  _Float16* k16    = (_Float16*)alloc((size_t)SEQ * NKV * HD * 2);
  _Float16* vT16   = (_Float16*)alloc((size_t)NKV * HD * SEQ * 2);
  _Float16* attn16 = (_Float16*)alloc((size_t)SEQ * HID * 2);
  float*    hid2   = (float*)alloc((size_t)SEQ * HID * 4);
  _Float16* h2n    = (_Float16*)alloc((size_t)SEQ * HID * 2);
  float*    gu     = (float*)alloc((size_t)SEQ * GUP * 4);
  _Float16* act16  = (_Float16*)alloc((size_t)SEQ * INTER * 2);

  // one-time (per launch) weight convert + transpose -> f16 [N][K]
  transpose_f32_to_f16<<<dim3(NQKV / 32, HID / 32), dim3(32, 8), 0, stream>>>(
      w_qkv, wqkvT, HID, NQKV);
  transpose_f32_to_f16<<<dim3(HID / 32, HID / 32), dim3(32, 8), 0, stream>>>(
      w_o, woT, HID, HID);
  transpose_f32_to_f16<<<dim3(GUP / 32, HID / 32), dim3(32, 8), 0, stream>>>(
      w_gate_up, wguT, HID, GUP);
  transpose_f32_to_f16<<<dim3(HID / 32, INTER / 32), dim3(32, 8), 0, stream>>>(
      w_down, wdnT, INTER, HID);

  // attention block
  rmsnorm_f16_kernel<<<SEQ, 256, 0, stream>>>(hidden, ln1_w, h1n, HID);
  gemm_wmma_f16<<<dim3(SEQ / 32, NQKV / 512), 256, 0, stream>>>(
      h1n, wqkvT, nullptr, qkv, SEQ, NQKV, HID);
  rope_split_kernel<<<SEQ, 256, 0, stream>>>(positions, qkv, q16, k16, vT16);
  flash_attn_kernel<<<dim3(SEQ / 16, NH), 32, 0, stream>>>(q16, k16, vT16,
                                                           attn16);
  gemm_wmma_f16<<<dim3(SEQ / 32, HID / 512), 256, 0, stream>>>(
      attn16, woT, hidden, hid2, SEQ, HID, HID);

  // MLP block
  rmsnorm_f16_kernel<<<SEQ, 256, 0, stream>>>(hid2, ln2_w, h2n, HID);
  gemm_wmma_f16<<<dim3(SEQ / 32, GUP / 512), 256, 0, stream>>>(
      h2n, wguT, nullptr, gu, SEQ, GUP, HID);
  {
    size_t total = (size_t)SEQ * INTER;
    silu_mul_kernel<<<(unsigned)((total + 255) / 256), 256, 0, stream>>>(gu,
                                                                         act16);
  }
  gemm_wmma_f16<<<dim3(SEQ / 32, HID / 512), 256, 0, stream>>>(
      act16, wdnT, hid2, out, SEQ, HID, INTER);
}
